// MinkUNetBase_73023033967245
// MI455X (gfx1250) — matrix operand order repaired and checked by reading, not compiled, not run on hardware
//
#include <hip/hip_runtime.h>
#include <hip/hip_bf16.h>

// ---------------------------------------------------------------------------
// CDNA5 (gfx1250) implicit-GEMM 3D conv via V_WMMA_F32_16X16X32_BF16, wave32.
// Each wave: 1 N-tile (16 output voxels) x MT M-tiles (16*MT output channels),
// B fragment reused across MT WMMAs per K-step of 32.
// ---------------------------------------------------------------------------

typedef __attribute__((ext_vector_type(16))) __bf16          v16bf;
typedef __attribute__((ext_vector_type(16))) unsigned short  v16u;
typedef __attribute__((ext_vector_type(8)))  float           v8f;

union ABFrag { v16u u; v16bf b; };

__device__ __forceinline__ unsigned short f32_to_bf16_bits(float f) {
  return (unsigned short)(__float_as_uint(f) >> 16);
}

#ifndef __has_builtin
#define __has_builtin(x) 0
#endif

#if __has_builtin(__builtin_amdgcn_global_load_async_to_lds_b32)
#define HAS_ASYNC_LDS 1
#else
#define HAS_ASYNC_LDS 0
#endif

#if HAS_ASYNC_LDS
// Builtin signature (per hipcc diagnostic): AS1 int* global src, AS3 int* LDS
// dst, i32 imm offset, i32 cpol.  Pointers built by integer round-trip (low 32
// bits of a flat LDS address are the LDS byte address, ISA 10.2).
typedef __attribute__((address_space(1))) int* as1_ptr;
typedef __attribute__((address_space(3))) int* as3_ptr;

__device__ __forceinline__ void wait_async0() {
#if __has_builtin(__builtin_amdgcn_s_wait_asynccnt)
  __builtin_amdgcn_s_wait_asynccnt(0);
#else
  asm volatile("s_wait_asynccnt 0x0" ::: "memory");
#endif
}
#endif

#define NWAVES 8   // 8 waves * 32 lanes = 256 threads; 8 N-tiles of 16 per block

// mode 0: standard conv, NCDHW x, OIDHW w, stride/pad.  Reduction R = Cin*k^3.
// mode 1: transpose conv k=4 s=4 VALID: blockIdx.z = kernel offset (dz,dy,dx),
//         N tiles over INPUT voxels, out[4*zi+dz,...], R = Cin, flipped weights.
template <int MT>
__global__ __launch_bounds__(256)
void conv3d_wmma(const float* __restrict__ x, const float* __restrict__ w,
                 float* __restrict__ y,
                 int Cin, int Cout,
                 int Din, int Hin, int Win,
                 int Dout, int Hout, int Wout,
                 int ks, int stride, int pad, int mode)
{
  __shared__ float          As32[MT * 16 * 32];     // A tile: (16*MT) cout x 32 r (f32)
  __shared__ unsigned short Bs[NWAVES][32 * 16];    // per-wave B tile: 32 r x 16 n (bf16)

  const int tid    = threadIdx.x;
  const int lane   = tid & 31;
  const int wave   = tid >> 5;
  const int coBase = (int)blockIdx.y * (16 * MT);
  const int kvol   = ks * ks * ks;
  const int R      = (mode == 0) ? Cin * kvol : Cin;
  const int Ntile  = (mode == 0) ? Dout * Hout * Wout : Din * Hin * Win;
  const int nBase  = ((int)blockIdx.x * NWAVES + wave) << 4;

  int dz = 0, dy = 0, dx = 0, woff = 0;
  if (mode) {
    const int off = (int)blockIdx.z;
    dz = off >> 4; dy = (off >> 2) & 3; dx = off & 3;
    woff = (3 - dz) * 16 + (3 - dy) * 4 + (3 - dx);   // spatial flip (conv_transpose)
  }

  // ---- per-lane N decomposition: loop-invariant across the whole K loop ----
  const int  nn     = lane & 15;
  const int  ni     = nBase + nn;
  const bool nvalid = ni < Ntile;
  int od = 0, oh = 0, ow = 0, zi = 0, yi = 0, xi = 0;
  size_t xsp = 0;
  if (mode == 0) {
    ow = ni % Wout; const int t = ni / Wout; oh = t % Hout; od = t / Hout;
  } else {
    xi = ni % Win;  const int t = ni / Win;  yi = t % Hin;  zi = t / Hin;
    xsp = ((size_t)zi * Hin + yi) * Win + xi;
  }
  const int odS = od * stride - pad;
  const int ohS = oh * stride - pad;
  const int owS = ow * stride - pad;
  const size_t Sin = (size_t)Din * Hin * Win;

  v8f acc[MT] = {};

  for (int r0 = 0; r0 < R; r0 += 32) {
    __syncthreads();

    // ---- stage A tile (weights) -----------------------------------------
#if HAS_ASYNC_LDS
    const bool fullA = (mode == 0) && (r0 + 32 <= R) && (coBase + 16 * MT <= Cout);
    if (fullA) {
      // Async global->LDS copy (ASYNCcnt-tracked), raw f32; convert on read.
      #pragma unroll
      for (int i = tid; i < MT * 512; i += 256) {
        const int mi = i >> 5;
        const int ri = r0 + (i & 31);
        const float* gp = w + (size_t)(coBase + mi) * R + ri;
        __builtin_amdgcn_global_load_async_to_lds_b32(
            (as1_ptr)(size_t)gp,
            (as3_ptr)(unsigned)(size_t)&As32[i],
            0, 0);
      }
    } else
#endif
    {
      #pragma unroll
      for (int i = tid; i < MT * 512; i += 256) {
        const int mi = i >> 5;
        const int ri = r0 + (i & 31);
        const int co = coBase + mi;
        float v = 0.f;
        if (co < Cout && ri < R)
          v = (mode == 0) ? w[(size_t)co * R + ri]
                          : w[((size_t)co * Cin + ri) * kvol + woff];
        As32[i] = v;
      }
    }

    if (r0 + 32 < R) __builtin_prefetch(w + (size_t)coBase * R + r0 + 32, 0, 1);

    // ---- stage per-wave B tile (input patches, zero-padded) --------------
    {
      const int rb = lane >> 4;
      #pragma unroll
      for (int k = 0; k < 16; ++k) {
        const int rl = rb + 2 * k;            // row 0..31 of the K-slice
        const int ri = r0 + rl;
        float v = 0.f;
        if (nvalid && ri < R) {
          if (mode == 0) {
            const int ci = ri / kvol;
            const int kk = ri % kvol;
            const int kx = kk % ks;
            const int t  = kk / ks;
            const int ky = t % ks;
            const int kz = t / ks;
            const int az = odS + kz, ay = ohS + ky, ax = owS + kx;
            if ((unsigned)az < (unsigned)Din && (unsigned)ay < (unsigned)Hin &&
                (unsigned)ax < (unsigned)Win)
              v = x[(((size_t)ci * Din + az) * Hin + ay) * Win + ax];
          } else {
            v = x[(size_t)ri * Sin + xsp];    // no divides: hoisted xsp
          }
        }
        Bs[wave][rl * 16 + nn] = f32_to_bf16_bits(v);
      }
    }

#if HAS_ASYNC_LDS
    wait_async0();
#endif
    __syncthreads();

    // ---- build fragments per ISA wave32 VGPR layouts, MT WMMAs -----------
    ABFrag Bf;
    {
      const int kb2 = (lane >> 4) << 3;       // 0 or 8
      #pragma unroll
      for (int j = 0; j < 8; ++j) {
        Bf.u[2 * j]     = Bs[wave][(kb2 + (j >> 2) * 16 + 2 * (j & 3)) * 16 + nn];
        Bf.u[2 * j + 1] = Bs[wave][(kb2 + (j >> 2) * 16 + 2 * (j & 3) + 1) * 16 + nn];
      }
    }
    const int mrow = lane & 15;
    const int kb   = (lane >> 4) << 3;        // 0 or 8
    #pragma unroll
    for (int mt = 0; mt < MT; ++mt) {
      ABFrag Af;
      const float* arow = &As32[(mt * 16 + mrow) * 32];
      #pragma unroll
      for (int j = 0; j < 4; ++j) {
        Af.u[2 * j]     = f32_to_bf16_bits(arow[kb + 2 * j]);          // VGPR j lo/hi
        Af.u[2 * j + 1] = f32_to_bf16_bits(arow[kb + 2 * j + 1]);
        Af.u[2 * j + 8] = f32_to_bf16_bits(arow[16 + kb + 2 * j]);     // VGPR 4+j
        Af.u[2 * j + 9] = f32_to_bf16_bits(arow[16 + kb + 2 * j + 1]);
      }
      acc[mt] = __builtin_amdgcn_wmma_f32_16x16x32_bf16(
          false, Af.b, false, Bf.b, (short)0, acc[mt], false, false);
    }
  }

  // ---- epilogue: scatter MT 16x16 f32 fragments to y ---------------------
  if (nvalid) {
    size_t outSpatial;
    if (mode == 0) {
      outSpatial = (size_t)ni;
    } else {
      outSpatial = ((size_t)(4 * zi + dz) * Hout + (4 * yi + dy)) * Wout + (4 * xi + dx);
    }
    const size_t S = (size_t)Dout * Hout * Wout;
    const int mAdd = (lane >> 4) << 3;
    #pragma unroll
    for (int mt = 0; mt < MT; ++mt) {
      #pragma unroll
      for (int r = 0; r < 8; ++r) {
        const int co = coBase + mt * 16 + mAdd + r;
        if (co < Cout) y[(size_t)co * S + outSpatial] = acc[mt][r];
      }
    }
  }
}

// ---------------------------------------------------------------------------
// Training-mode batch norm: per-channel mean/var -> scale/shift.
// ---------------------------------------------------------------------------
__global__ __launch_bounds__(256)
void bn_stats(const float* __restrict__ x, const float* __restrict__ g,
              const float* __restrict__ b, float* __restrict__ scale,
              float* __restrict__ shift, int C, size_t S)
{
  __shared__ float s1[256];
  __shared__ float s2[256];
  const int c = (int)blockIdx.x;
  const float* p = x + (size_t)c * S;
  float sum = 0.f, sq = 0.f;
  for (size_t i = threadIdx.x; i < S; i += 256) {
    const float v = p[i];
    sum += v; sq += v * v;
  }
  s1[threadIdx.x] = sum; s2[threadIdx.x] = sq;
  __syncthreads();
  for (int st = 128; st > 0; st >>= 1) {
    if ((int)threadIdx.x < st) {
      s1[threadIdx.x] += s1[threadIdx.x + st];
      s2[threadIdx.x] += s2[threadIdx.x + st];
    }
    __syncthreads();
  }
  if (threadIdx.x == 0) {
    const float inv = 1.f / (float)S;
    const float m   = s1[0] * inv;
    const float var = s2[0] * inv - m * m;
    const float sc  = g[c] * rsqrtf(var + 1e-5f);
    scale[c] = sc;
    shift[c] = b[c] - m * sc;
  }
  (void)C;
}

// mode bit0 = relu, bit1 = add skip.  y = [relu]( x*scale[c]+shift[c] [+skip] )
__global__ __launch_bounds__(256)
void bn_apply(const float* __restrict__ x, const float* __restrict__ skip,
              float* __restrict__ y, const float* __restrict__ scale,
              const float* __restrict__ shift, int C, size_t S, int mode)
{
  const size_t i = (size_t)blockIdx.x * 256 + threadIdx.x;
  const size_t total = (size_t)C * S;
  if (i >= total) return;
  const int c = (int)(i / S);
  float v = x[i] * scale[c] + shift[c];
  if (mode & 2) v += skip[i];
  if (mode & 1) v = fmaxf(v, 0.f);
  y[i] = v;
}

__global__ __launch_bounds__(256)
void copyk(const float* __restrict__ src, float* __restrict__ dst, size_t n)
{
  const size_t i = (size_t)blockIdx.x * 256 + threadIdx.x;
  if (i < n) dst[i] = src[i];
}

__global__ __launch_bounds__(256)
void bias_add(const float* __restrict__ x, const float* __restrict__ bias,
              float* __restrict__ y, int C, size_t S)
{
  const size_t i = (size_t)blockIdx.x * 256 + threadIdx.x;
  const size_t total = (size_t)C * S;
  if (i >= total) return;
  const int c = (int)(i / S);
  y[i] = x[i] + bias[c];
}

// ---------------------------------------------------------------------------
// Host orchestration of the full MinkUNet forward pass.
// ---------------------------------------------------------------------------
extern "C" void kernel_launch(void* const* d_in, const int* in_sizes, int n_in,
                              void* d_out, int out_size, void* d_ws, size_t ws_size,
                              hipStream_t stream)
{
  (void)in_sizes; (void)n_in; (void)out_size; (void)ws_size;

  int pi = 0;
  auto nxt = [&]() -> const float* { return (const float*)d_in[pi++]; };
  const float* x_in = nxt();   // (1, 3, 64, 64, 64)

  const size_t S64 = 64ull * 64 * 64;
  const size_t S16 = 16ull * 16 * 16;
  const size_t S4  = 4ull * 4 * 4;

  float* W = (float*)d_ws;
  size_t woff = 0;
  auto alloc = [&](size_t n) { float* p = W + woff; woff += n; return p; };

  float* d_scale = alloc(256);
  float* d_shift = alloc(256);
  float* P1   = alloc(32 * S64);    // out_p1   (skip, 32 @ 64^3)
  float* B1P2 = alloc(32 * S16);    // out_b1p2 (skip, 32 @ 16^3)
  float* B2P4 = alloc(64 * S4);     // out_b2p4 (skip, 64 @ 4^3)
  float* M0 = alloc(192 * S16);     // small-stage slots (<= 16^3)
  float* M1 = alloc(192 * S16);
  float* M2 = alloc(192 * S16);
  float* M3 = alloc(192 * S16);
  float* G0 = alloc(128 * S64);     // big-stage slots (64^3)
  float* G1 = alloc(128 * S64);
  float* G2 = alloc(128 * S64);
  float* G3 = alloc(128 * S64);

  float* OUT    = (float*)d_out;
  float* CODE   = OUT;              // out_code: 128 floats
  float* LOGITS = OUT + 128;        // logits:   20 * 64^3 floats

  auto conv = [&](const float* xin, const float* w, float* y, int Cin, int Cout,
                  int Din, int Hin, int Win, int Dout, int Hout, int Wout,
                  int ks, int stride, int pad, int mode) {
    const int Ntile = (mode == 0) ? Dout * Hout * Wout : Din * Hin * Win;
    const int MT = (Cout % 64 == 0) ? 4 : (Cout % 48 == 0) ? 3
                 : (Cout % 32 == 0) ? 2 : 1;
    dim3 grid((unsigned)((Ntile + 127) / 128),
              (unsigned)((Cout + 16 * MT - 1) / (16 * MT)),
              mode ? 64u : 1u);
    switch (MT) {
      case 4: conv3d_wmma<4><<<grid, dim3(256), 0, stream>>>(
                xin, w, y, Cin, Cout, Din, Hin, Win, Dout, Hout, Wout,
                ks, stride, pad, mode); break;
      case 3: conv3d_wmma<3><<<grid, dim3(256), 0, stream>>>(
                xin, w, y, Cin, Cout, Din, Hin, Win, Dout, Hout, Wout,
                ks, stride, pad, mode); break;
      case 2: conv3d_wmma<2><<<grid, dim3(256), 0, stream>>>(
                xin, w, y, Cin, Cout, Din, Hin, Win, Dout, Hout, Wout,
                ks, stride, pad, mode); break;
      default: conv3d_wmma<1><<<grid, dim3(256), 0, stream>>>(
                xin, w, y, Cin, Cout, Din, Hin, Win, Dout, Hout, Wout,
                ks, stride, pad, mode); break;
    }
  };

  auto bn = [&](float* xin, float* y, const float* skip, const float* g,
                const float* b, int C, size_t S, int mode) {
    bn_stats<<<dim3((unsigned)C), dim3(256), 0, stream>>>(xin, g, b, d_scale,
                                                          d_shift, C, S);
    const size_t tot = (size_t)C * S;
    bn_apply<<<dim3((unsigned)((tot + 255) / 256)), dim3(256), 0, stream>>>(
        xin, skip, y, d_scale, d_shift, C, S, mode);
  };

  auto copy = [&](const float* src, float* dst, size_t n) {
    copyk<<<dim3((unsigned)((n + 255) / 256)), dim3(256), 0, stream>>>(src, dst, n);
  };

  // BasicBlock: relu(bn2(conv2(relu(bn1(conv1(x))))) + shortcut(x))
  auto basic_block = [&](const float* xin, int Cin, int Cout,
                         int Dd, int Hh, int Ww, bool hasDw,
                         float* A, float* B, float* C, float* out) {
    const size_t S = (size_t)Dd * Hh * Ww;
    const float* w1 = nxt(); const float* w2 = nxt();
    const float* g1 = nxt(); const float* b1 = nxt();
    const float* g2 = nxt(); const float* b2 = nxt();
    const float* dw = nullptr; const float* dg = nullptr; const float* db = nullptr;
    if (hasDw) { dw = nxt(); dg = nxt(); db = nxt(); }

    conv(xin, w1, A, Cin, Cout, Dd, Hh, Ww, Dd, Hh, Ww, 3, 1, 1, 0);
    bn(A, A, nullptr, g1, b1, Cout, S, 1);                       // bn1 + relu
    conv(A, w2, B, Cout, Cout, Dd, Hh, Ww, Dd, Hh, Ww, 3, 1, 1, 0);
    const float* skip = xin;
    if (hasDw) {
      conv(xin, dw, C, Cin, Cout, Dd, Hh, Ww, Dd, Hh, Ww, 1, 1, 0, 0);
      bn(C, C, nullptr, dg, db, Cout, S, 0);                     // bn shortcut
      skip = C;
    }
    bn(B, out, skip, g2, b2, Cout, S, 3);                        // bn2 + add + relu
  };

  // ---- encoder -----------------------------------------------------------
  { const float* w = nxt();                                      // conv0 5^3 SAME
    conv(x_in, w, P1, 3, 32, 64, 64, 64, 64, 64, 64, 5, 1, 2, 0);
    const float* g = nxt(); const float* b = nxt();
    bn(P1, P1, nullptr, g, b, 32, S64, 1); }                     // out_p1

  { const float* w = nxt();                                      // conv1 4^3 s4
    conv(P1, w, M0, 32, 32, 64, 64, 64, 16, 16, 16, 4, 4, 0, 0);
    const float* g = nxt(); const float* b = nxt();
    bn(M0, M0, nullptr, g, b, 32, S16, 1); }

  basic_block(M0, 32, 32, 16, 16, 16, false, M1, M2, M3, M3);    // block1[0]
  basic_block(M3, 32, 32, 16, 16, 16, false, M1, M2, M0, B1P2);  // block1[1] -> out_b1p2

  { const float* w = nxt();                                      // conv2 4^3 s4
    conv(B1P2, w, M0, 32, 32, 16, 16, 16, 4, 4, 4, 4, 4, 0, 0);
    const float* g = nxt(); const float* b = nxt();
    bn(M0, M0, nullptr, g, b, 32, S4, 1); }

  basic_block(M0, 32, 64, 4, 4, 4, true,  M1, M2, M3, M0);       // block2[0]
  basic_block(M0, 64, 64, 4, 4, 4, false, M1, M2, M3, B2P4);     // block2[1] -> out_b2p4

  { const float* w = nxt();                                      // conv3 4^3 s4
    conv(B2P4, w, M0, 64, 64, 4, 4, 4, 1, 1, 1, 4, 4, 0, 0);
    const float* g = nxt(); const float* b = nxt();
    bn(M0, M0, nullptr, g, b, 64, 1, 1); }

  basic_block(M0, 64, 128, 1, 1, 1, true,  M1, M2, M3, M0);      // block3[0]
  basic_block(M0, 128, 128, 1, 1, 1, false, M1, M2, M3, CODE);   // block3[1] -> out_code

  // ---- decoder -----------------------------------------------------------
  { const float* w = nxt();                                      // convtr5 4x up
    conv(CODE, w, M0, 128, 128, 1, 1, 1, 4, 4, 4, 4, 4, 0, 1);
    const float* g = nxt(); const float* b = nxt();
    bn(M0, M0, nullptr, g, b, 128, S4, 1); }

  copy(M0, M1, 128 * S4); copy(B2P4, M1 + 128 * S4, 64 * S4);    // concat -> 192 @ 4^3
  basic_block(M1, 192, 128, 4, 4, 4, true,  M2, M3, M0, M1);     // block6[0]
  basic_block(M1, 128, 128, 4, 4, 4, false, M2, M3, M0, M0);     // block6[1]

  { const float* w = nxt();                                      // convtr6 4x up
    conv(M0, w, M1, 128, 96, 4, 4, 4, 16, 16, 16, 4, 4, 0, 1);
    const float* g = nxt(); const float* b = nxt();
    bn(M1, M1, nullptr, g, b, 96, S16, 1); }

  copy(M1, M2, 96 * S16); copy(B1P2, M2 + 96 * S16, 32 * S16);   // concat -> 128 @ 16^3
  basic_block(M2, 128, 96, 16, 16, 16, true,  M0, M1, M3, M2);   // block7[0]
  basic_block(M2, 96, 96, 16, 16, 16, false, M0, M1, M3, M3);    // block7[1]

  { const float* w = nxt();                                      // convtr7 4x up
    conv(M3, w, G0, 96, 96, 16, 16, 16, 64, 64, 64, 4, 4, 0, 1);
    const float* g = nxt(); const float* b = nxt();
    bn(G0, G0, nullptr, g, b, 96, S64, 1); }

  copy(G0, G1, 96 * S64); copy(P1, G1 + 96 * S64, 32 * S64);     // concat -> 128 @ 64^3
  basic_block(G1, 128, 96, 64, 64, 64, true,  G2, G3, G0, G1);   // block8[0]
  basic_block(G1, 96, 96, 64, 64, 64, false, G2, G3, G0, G0);    // block8[1]

  // ---- final 1x1x1 conv + bias ------------------------------------------
  { const float* w = nxt();
    conv(G0, w, G2, 96, 20, 64, 64, 64, 64, 64, 64, 1, 1, 0, 0);
    const float* b = nxt();
    const size_t tot = 20 * S64;
    bias_add<<<dim3((unsigned)((tot + 255) / 256)), dim3(256), 0, stream>>>(
        G2, b, LOGITS, 20, S64);
  }
}
